// Seq_Encoder_23897198035542
// MI455X (gfx1250) — matrix operand order repaired and checked by reading, not compiled, not run on hardware
//
#include <hip/hip_runtime.h>
#include <hip/hip_bf16.h>
#include <math.h>

typedef __attribute__((ext_vector_type(16))) _Float16 v16h;
typedef __attribute__((ext_vector_type(8)))  _Float16 v8h;
typedef __attribute__((ext_vector_type(4)))  _Float16 v4h;
typedef __attribute__((ext_vector_type(8)))  float    v8f;
typedef __attribute__((ext_vector_type(4)))  float    v4f;

#define Bc   4
#define Sc   1024
#define Dc   512
#define Hc   8
#define Cc   64
#define MLPc 2048
#define Lc   4
#define ROWS (Bc * Sc)          // 4096
#define TOPKc 256

// ---------------------------------------------------------------------------
// WMMA helpers (layouts per CDNA5 ISA 7.12.2, wave32) — all fragment loads are
// contiguous 16B vector loads per lane.
// ---------------------------------------------------------------------------
__device__ __forceinline__ v8f wmma_f16(v16h a, v16h b, v8f c) {
  return __builtin_amdgcn_wmma_f32_16x16x32_f16(false, a, false, b, (short)0, c,
                                                false, false);
}

__device__ __forceinline__ v16h cat8(v8h a, v8h b) {
  return __builtin_shufflevector(a, b, 0, 1, 2, 3, 4, 5, 6, 7, 8, 9, 10, 11, 12,
                                 13, 14, 15);
}

// A-frag 16x32 f16: source row-major [m][k], stride lda halves.
// Per lane: runs r[0..7] and r[16..23] with r = row + 8*hi.
__device__ __forceinline__ v16h load_a16(const _Float16* A, int lda) {
  int l = threadIdx.x & 31, m = l & 15, hi = l >> 4;
  const _Float16* r = A + m * lda + 8 * hi;
  v8h lo = *(const v8h*)r;
  v8h hv = *(const v8h*)(r + 16);
  return cat8(lo, hv);
}

// B-frag 32x16 f16 where the logical B[k][n] is stored transposed as Bt[n][k]
// (stride ldb halves). Per lane: runs r[0..7] and r[16..23] with r = Bt[n] + 8*hi.
__device__ __forceinline__ v16h load_bt16(const _Float16* Bt, int ldb) {
  int l = threadIdx.x & 31, n = l & 15, hi = l >> 4;
  const _Float16* r = Bt + n * ldb + 8 * hi;
  v8h x0 = *(const v8h*)r;
  v8h x1 = *(const v8h*)(r + 16);
  return cat8(x0, x1);
}

// A-frag 16x32 from f32 source (softmax probs in LDS), stride lda floats.
__device__ __forceinline__ v16h load_a32(const float* A, int lda) {
  int l = threadIdx.x & 31, m = l & 15, hi = l >> 4;
  const float* r = A + m * lda + 8 * hi;
  v4f f0 = *(const v4f*)r;
  v4f f1 = *(const v4f*)(r + 4);
  v4f f2 = *(const v4f*)(r + 16);
  v4f f3 = *(const v4f*)(r + 20);
  v16h a;
#pragma unroll
  for (int j = 0; j < 4; ++j) {
    a[j]      = (_Float16)f0[j];
    a[4 + j]  = (_Float16)f1[j];
    a[8 + j]  = (_Float16)f2[j];
    a[12 + j] = (_Float16)f3[j];
  }
  return a;
}

// ---------------------------------------------------------------------------
// Order-preserving float <-> uint key (radix top-k select)
// ---------------------------------------------------------------------------
__device__ __forceinline__ unsigned f2key(float f) {
  unsigned u = __float_as_uint(f);
  return (u & 0x80000000u) ? ~u : (u | 0x80000000u);
}
__device__ __forceinline__ float key2f(unsigned k) {
  unsigned u = (k & 0x80000000u) ? (k & 0x7FFFFFFFu) : ~k;
  return __uint_as_float(u);
}

// ---------------------------------------------------------------------------
// Elementwise / setup kernels
// ---------------------------------------------------------------------------
__global__ void addpos_kernel(const float* __restrict__ x,
                              const float* __restrict__ pos,
                              float* __restrict__ X) {
  int idx = blockIdx.x * blockDim.x + threadIdx.x;
  if (idx >= Bc * Sc * Dc) return;
  X[idx] = x[idx] + pos[idx & (Sc * Dc - 1)];
}

// Plain f32 -> f16 convert, 4 elements/thread (weights keep [O][Dk] layout,
// which is exactly the transposed-B layout the fragment loader wants).
__global__ void wcvt_kernel(const float* __restrict__ W,
                            _Float16* __restrict__ T, int n) {
  int idx = (blockIdx.x * blockDim.x + threadIdx.x) * 4;
  if (idx >= n) return;
  v4f wv = *(const v4f*)(W + idx);
  v4h o;
#pragma unroll
  for (int j = 0; j < 4; ++j) o[j] = (_Float16)wv[j];
  *(v4h*)(T + idx) = o;
}

// LayerNorm over D=512, one block (256 thr) per row.
__global__ void ln_kernel(const float* __restrict__ X,
                          const float* __restrict__ w,
                          const float* __restrict__ bprm,
                          _Float16* __restrict__ out16,
                          float* __restrict__ out32) {
  __shared__ float red[256];
  __shared__ float stat[2];
  int row = blockIdx.x, t = threadIdx.x;
  const float* xr = X + (size_t)row * Dc;
  float x0 = xr[t], x1 = xr[t + 256];
  red[t] = x0 + x1;
  __syncthreads();
  for (int s = 128; s > 0; s >>= 1) {
    if (t < s) red[t] += red[t + s];
    __syncthreads();
  }
  if (t == 0) stat[0] = red[0] * (1.0f / (float)Dc);
  __syncthreads();
  float mu = stat[0];
  float d0 = x0 - mu, d1 = x1 - mu;
  red[t] = d0 * d0 + d1 * d1;
  __syncthreads();
  for (int s = 128; s > 0; s >>= 1) {
    if (t < s) red[t] += red[t + s];
    __syncthreads();
  }
  if (t == 0) stat[1] = rsqrtf(red[0] * (1.0f / (float)Dc) + 1e-6f);
  __syncthreads();
  float rstd = stat[1];
  float y0 = d0 * rstd * w[t] + bprm[t];
  float y1 = d1 * rstd * w[t + 256] + bprm[t + 256];
  if (out16) {
    out16[(size_t)row * Dc + t] = (_Float16)y0;
    out16[(size_t)row * Dc + t + 256] = (_Float16)y1;
  } else {
    out32[(size_t)row * Dc + t] = y0;
    out32[(size_t)row * Dc + t + 256] = y1;
  }
}

// Depthwise conv3 along S + head split.
// qkv: [ROWS][3D] f32 (row = b*S+s).
// q,k -> f32 (B,H,S,C) row-major; v -> f16 TRANSPOSED (B,H,C,S).
__global__ void dwconv_split_kernel(const float* __restrict__ qkv,
                                    const float* __restrict__ dw,
                                    float* __restrict__ qf,
                                    float* __restrict__ kf,
                                    _Float16* __restrict__ vT) {
  int idx = blockIdx.x * blockDim.x + threadIdx.x;
  if (idx >= ROWS * 3 * Dc) return;
  int o = idx % (3 * Dc);
  int row = idx / (3 * Dc);
  int s = row & (Sc - 1);
  int b = row >> 10;
  const float* col = qkv + (size_t)row * (3 * Dc) + o;
  float c0 = (s > 0) ? col[-(3 * Dc)] : 0.0f;
  float c1 = col[0];
  float c2 = (s < Sc - 1) ? col[3 * Dc] : 0.0f;
  const float* wr = dw + (size_t)o * 3;
  float y = wr[0] * c0 + wr[1] * c1 + wr[2] * c2;
  int which = o >> 9;          // 0=q 1=k 2=v
  int hc = o & (Dc - 1);       // h*64 + c
  int bh = b * Hc + (hc >> 6);
  int c = hc & 63;
  if (which == 0) qf[(((size_t)bh * Sc + s) << 6) + c] = y;
  else if (which == 1) kf[(((size_t)bh * Sc + s) << 6) + c] = y;
  else vT[((size_t)bh * Cc + c) * Sc + s] = (_Float16)y;
}

// L2-normalize 64-wide rows (one wave per row), f32 -> f16 [s][c].
__global__ void l2norm_kernel(const float* __restrict__ in,
                              _Float16* __restrict__ out, int nrows) {
  int wid = (blockIdx.x * blockDim.x + threadIdx.x) >> 5;
  int lane = threadIdx.x & 31;
  if (wid >= nrows) return;
  const float* r = in + (size_t)wid * 64;
  float a = r[lane], b = r[lane + 32];
  float ss = a * a + b * b;
  for (int off = 16; off; off >>= 1) ss += __shfl_xor(ss, off);
  float sc = 1.0f / fmaxf(sqrtf(ss), 1e-12f);
  out[(size_t)wid * 64 + lane] = (_Float16)(a * sc);
  out[(size_t)wid * 64 + lane + 32] = (_Float16)(b * sc);
}

// ---------------------------------------------------------------------------
// WMMA GEMM, LDS-free: C[M][N] = A[M][K] (f16 rm) @ Bt[N][K] (f16, = W layout)
// mode 0: out32 = acc ; 1: out32 += acc ; 2: out16 = gelu(acc+bias) ;
// mode 3: out32 += acc + bias
// Block: 256 thr = 8 waves; tile 32x128; each wave owns 1 row-tile x 2 col-tiles.
// ---------------------------------------------------------------------------
#define GMT 32
#define GNT 128
__global__ void gemm_kernel(const _Float16* __restrict__ A,
                            const _Float16* __restrict__ Bt,
                            int M, int N, int K,
                            const float* __restrict__ bias,
                            float* __restrict__ out32,
                            _Float16* __restrict__ out16, int mode) {
  int m0 = blockIdx.x * GMT, n0 = blockIdx.y * GNT;
  int w = threadIdx.x >> 5, lane = threadIdx.x & 31;
  int rt = w & 1;
  int ctb = (w >> 1) * 2;
  const _Float16* Ab = A + (size_t)(m0 + rt * 16) * K;
  const _Float16* B0 = Bt + (size_t)(n0 + ctb * 16) * K;
  const _Float16* B1 = Bt + (size_t)(n0 + (ctb + 1) * 16) * K;
  v8f acc0 = {};
  v8f acc1 = {};
  for (int kc = 0; kc < K; kc += 32) {
    v16h a  = load_a16(Ab + kc, K);
    v16h b0 = load_bt16(B0 + kc, K);
    v16h b1 = load_bt16(B1 + kc, K);
    acc0 = wmma_f16(a, b0, acc0);
    acc1 = wmma_f16(a, b1, acc1);
  }
  int hi = lane >> 4, nn = lane & 15;
#pragma unroll
  for (int tt = 0; tt < 2; ++tt) {
    v8f acc = tt ? acc1 : acc0;
    int col = n0 + (ctb + tt) * 16 + nn;
    float bv = bias ? bias[col] : 0.0f;
#pragma unroll
    for (int i = 0; i < 8; ++i) {
      size_t off = (size_t)(m0 + rt * 16 + i + 8 * hi) * N + col;
      float v = acc[i];
      if (mode == 0) {
        out32[off] = v;
      } else if (mode == 1) {
        out32[off] += v;
      } else if (mode == 2) {
        float z = v + bv;
        out16[off] = (_Float16)(0.5f * z * (1.0f + erff(z * 0.70710678118654752f)));
      } else {
        out32[off] += v + bv;
      }
    }
  }
}

// ---------------------------------------------------------------------------
// Sparse top-k attention. 1 block per (b,h, 32-query tile); 256 thr = 8 waves.
// Dynamic LDS: scores f32[32][1024] | hist u32[8][256]
// q16,k16: (B,H,S,C) f16 row-major.  vT: (B,H,C,S) f16 (transposed).
// ---------------------------------------------------------------------------
#define QT 32
#define ATTN_LDS (QT * Sc * 4 + 8 * 256 * 4)
__global__ void attn_kernel(const _Float16* __restrict__ q16,
                            const _Float16* __restrict__ k16,
                            const _Float16* __restrict__ vT,
                            const float* __restrict__ temp,
                            _Float16* __restrict__ outA) {
  extern __shared__ char smraw[];
  float*    smS = (float*)smraw;                    // 32x1024
  unsigned* smH = (unsigned*)(smraw + QT * Sc * 4); // 8x256

  int blk = blockIdx.x;
  int qb = blk & 31;
  int bh = blk >> 5;
  int h = bh & (Hc - 1), b = bh >> 3;
  int q0 = qb * QT;
  int t = threadIdx.x, w = t >> 5, lane = t & 31;
  const _Float16* Qg = q16 + ((size_t)bh * Sc + q0) * Cc;
  const _Float16* Kg = k16 + (size_t)bh * Sc * Cc;
  const _Float16* Vt = vT + (size_t)bh * Cc * Sc;
  float tmp = temp[h];

  // ---- scores = (q @ k^T) * temp : wave w owns columns [w*128, w*128+128)
  {
    int colb = w * 128;
#pragma unroll
    for (int rt = 0; rt < 2; ++rt) {
      v16h a0 = load_a16(Qg + rt * 16 * Cc, Cc);       // c = 0..31
      v16h a1 = load_a16(Qg + rt * 16 * Cc + 32, Cc);  // c = 32..63
#pragma unroll
      for (int ct = 0; ct < 8; ++ct) {
        int cb = colb + ct * 16;
        v8f acc = {};
        v16h b0 = load_bt16(Kg + (size_t)cb * Cc, Cc);
        v16h b1 = load_bt16(Kg + (size_t)cb * Cc + 32, Cc);
        acc = wmma_f16(a0, b0, acc);
        acc = wmma_f16(a1, b1, acc);
        int hi = lane >> 4, nn = lane & 15;
#pragma unroll
        for (int i = 0; i < 8; ++i)
          smS[(rt * 16 + i + 8 * hi) * Sc + cb + nn] = acc[i] * tmp;
      }
    }
  }
  __syncthreads();

  // ---- per-row exact top-256 threshold (radix select) + softmax (f32)
  unsigned* hist = smH + w * 256;
  for (int rr = 0; rr < 4; ++rr) {   // wave w handles rows w*4 .. w*4+3
    int row = w * 4 + rr;
    float* rs = smS + row * Sc;
    unsigned prefix = 0;
    int want = TOPKc;
    for (int pass = 0; pass < 4; ++pass) {
      int shift = 24 - pass * 8;
      for (int i = lane; i < 256; i += 32) hist[i] = 0;
      __syncthreads();
      for (int j = lane; j < Sc; j += 32) {
        unsigned key = f2key(rs[j]);
        if (pass == 0 || (key >> (shift + 8)) == prefix)
          atomicAdd(&hist[(key >> shift) & 255u], 1u);
      }
      __syncthreads();
      int cum = 0;
#pragma unroll 1
      for (int bkt = 255; bkt >= 0; --bkt) {
        int c = (int)hist[bkt];
        if (cum + c >= want) {
          prefix = (prefix << 8) | (unsigned)bkt;
          want -= cum;
          break;
        }
        cum += c;
      }
      __syncthreads();
    }
    float thr = key2f(prefix);
    float mx = -3.0e38f;
    for (int j = lane; j < Sc; j += 32) {
      float s = rs[j];
      if (s >= thr) mx = fmaxf(mx, fminf(s, 10000.0f));
    }
    for (int off = 16; off; off >>= 1) mx = fmaxf(mx, __shfl_xor(mx, off));
    float sum = 0.0f;
    for (int j = lane; j < Sc; j += 32) {
      float s = rs[j];
      if (s >= thr) sum += expf(fminf(s, 10000.0f) - mx);
    }
    for (int off = 16; off; off >>= 1) sum += __shfl_xor(sum, off);
    float rinv = 1.0f / sum;
    for (int j = lane; j < Sc; j += 32) {
      float s = rs[j];
      rs[j] = (s >= thr) ? expf(fminf(s, 10000.0f) - mx) * rinv : 0.0f;
    }
    __syncthreads();
  }
  __syncthreads();

  // ---- out = probs @ V : wave w -> row tile (w>>2), col tile (w&3)
  {
    int rt = w >> 2, ct = w & 3;
    const _Float16* Vb = Vt + (size_t)(ct * 16) * Sc;
    v8f acc = {};
    for (int kc = 0; kc < Sc; kc += 32) {
      v16h a  = load_a32(smS + (rt * 16) * Sc + kc, Sc);
      v16h bb = load_bt16(Vb + kc, Sc);
      acc = wmma_f16(a, bb, acc);
    }
    int hi = lane >> 4, nn = lane & 15;
#pragma unroll
    for (int i = 0; i < 8; ++i) {
      int srow = q0 + rt * 16 + i + 8 * hi;
      outA[((size_t)(b * Sc + srow)) * Dc + h * Cc + ct * 16 + nn] =
          (_Float16)acc[i];
    }
  }
}

// ---------------------------------------------------------------------------
// Host orchestration
// ---------------------------------------------------------------------------
extern "C" void kernel_launch(void* const* d_in, const int* in_sizes, int n_in,
                              void* d_out, int out_size, void* d_ws,
                              size_t ws_size, hipStream_t stream) {
  const float* x_in   = (const float*)d_in[0];
  const float* pos    = (const float*)d_in[1];
  const float* ln1_w  = (const float*)d_in[2];
  const float* ln1_b  = (const float*)d_in[3];
  const float* qkv_w  = (const float*)d_in[4];
  const float* dw_w   = (const float*)d_in[5];
  const float* temp   = (const float*)d_in[6];
  const float* proj_w = (const float*)d_in[7];
  const float* ln2_w  = (const float*)d_in[8];
  const float* ln2_b  = (const float*)d_in[9];
  const float* mlp_w1 = (const float*)d_in[10];
  const float* mlp_b1 = (const float*)d_in[11];
  const float* mlp_w2 = (const float*)d_in[12];
  const float* mlp_b2 = (const float*)d_in[13];
  const float* lnf_w  = (const float*)d_in[14];
  const float* lnf_b  = (const float*)d_in[15];
  float* out = (float*)d_out;

  char* ws = (char*)d_ws;
  size_t off = 0;
  auto carve = [&](size_t bytes) {
    char* p = ws + off;
    off += (bytes + 255) & ~(size_t)255;
    return p;
  };
  float*    X      = (float*)carve((size_t)ROWS * Dc * 4);
  _Float16* hn16   = (_Float16*)carve((size_t)ROWS * Dc * 2);
  float*    qkvbuf = (float*)carve((size_t)ROWS * 3 * Dc * 4);
  float*    qf32   = (float*)carve((size_t)Bc * Hc * Sc * Cc * 4);
  float*    kf32   = (float*)carve((size_t)Bc * Hc * Sc * Cc * 4);
  _Float16* q16    = (_Float16*)carve((size_t)Bc * Hc * Sc * Cc * 2);
  _Float16* k16    = (_Float16*)carve((size_t)Bc * Hc * Sc * Cc * 2);
  _Float16* vT     = (_Float16*)carve((size_t)Bc * Hc * Sc * Cc * 2);
  _Float16* attn16 = (_Float16*)carve((size_t)ROWS * Dc * 2);
  _Float16* hid16  = (_Float16*)carve((size_t)ROWS * MLPc * 2);
  _Float16* qkvT   = (_Float16*)carve((size_t)Lc * 3 * Dc * Dc * 2);
  _Float16* projT  = (_Float16*)carve((size_t)Lc * Dc * Dc * 2);
  _Float16* w1T    = (_Float16*)carve((size_t)Lc * MLPc * Dc * 2);
  _Float16* w2T    = (_Float16*)carve((size_t)Lc * Dc * MLPc * 2);
  (void)ws_size; (void)n_in; (void)in_sizes; (void)out_size;

  // f32 -> f16 weight conversion (keeps [O][Dk] layout = transposed-B layout)
  {
    int n;
    n = Lc * 3 * Dc * Dc;
    wcvt_kernel<<<(n / 4 + 255) / 256, 256, 0, stream>>>(qkv_w, qkvT, n);
    n = Lc * Dc * Dc;
    wcvt_kernel<<<(n / 4 + 255) / 256, 256, 0, stream>>>(proj_w, projT, n);
    n = Lc * MLPc * Dc;
    wcvt_kernel<<<(n / 4 + 255) / 256, 256, 0, stream>>>(mlp_w1, w1T, n);
    n = Lc * Dc * MLPc;
    wcvt_kernel<<<(n / 4 + 255) / 256, 256, 0, stream>>>(mlp_w2, w2T, n);
  }

  // X = x + pos_emb
  {
    int n = ROWS * Dc;
    addpos_kernel<<<(n + 255) / 256, 256, 0, stream>>>(x_in, pos, X);
  }

  for (int l = 0; l < Lc; ++l) {
    // LN1 -> hn16
    ln_kernel<<<ROWS, 256, 0, stream>>>(X, ln1_w + l * Dc, ln1_b + l * Dc,
                                        hn16, nullptr);
    // QKV GEMM: [4096x512] @ [512x1536] -> f32
    gemm_kernel<<<dim3(ROWS / GMT, (3 * Dc) / GNT), 256, 0, stream>>>(
        hn16, qkvT + (size_t)l * 3 * Dc * Dc, ROWS, 3 * Dc, Dc, nullptr,
        qkvbuf, nullptr, 0);
    // depthwise conv3 + head split (+ V transpose)
    {
      int n = ROWS * 3 * Dc;
      dwconv_split_kernel<<<(n + 255) / 256, 256, 0, stream>>>(
          qkvbuf, dw_w + (size_t)l * 3 * Dc * 3, qf32, kf32, vT);
    }
    // L2 normalize q,k -> f16
    {
      int nrows = Bc * Hc * Sc;
      int blocks = (nrows * 32 + 255) / 256;
      l2norm_kernel<<<blocks, 256, 0, stream>>>(qf32, q16, nrows);
      l2norm_kernel<<<blocks, 256, 0, stream>>>(kf32, k16, nrows);
    }
    // sparse top-k attention
    attn_kernel<<<Bc * Hc * (Sc / QT), 256, ATTN_LDS, stream>>>(
        q16, k16, vT, temp + l * Hc, attn16);
    // proj GEMM, residual accumulate into X
    gemm_kernel<<<dim3(ROWS / GMT, Dc / GNT), 256, 0, stream>>>(
        attn16, projT + (size_t)l * Dc * Dc, ROWS, Dc, Dc, nullptr, X, nullptr,
        1);
    // LN2 -> hn16
    ln_kernel<<<ROWS, 256, 0, stream>>>(X, ln2_w + l * Dc, ln2_b + l * Dc,
                                        hn16, nullptr);
    // MLP1: bias + exact GELU -> f16 hidden
    gemm_kernel<<<dim3(ROWS / GMT, MLPc / GNT), 256, 0, stream>>>(
        hn16, w1T + (size_t)l * MLPc * Dc, ROWS, MLPc, Dc, mlp_b1 + l * MLPc,
        nullptr, hid16, 2);
    // MLP2: bias + residual accumulate into X
    gemm_kernel<<<dim3(ROWS / GMT, Dc / GNT), 256, 0, stream>>>(
        hid16, w2T + (size_t)l * Dc * MLPc, ROWS, Dc, MLPc, mlp_b2 + l * Dc, X,
        nullptr, 3);
  }

  // final LayerNorm -> f32 output
  ln_kernel<<<ROWS, 256, 0, stream>>>(X, lnf_w, lnf_b, nullptr, out);
}
